// LiquidEchoHead_75892072120998
// MI455X (gfx1250) — compile-verified, matrix operand order checked by reading, not run on hardware
//
#include <hip/hip_runtime.h>
#include <math.h>

// CDNA5 / gfx1250 wave32 WMMA types
typedef __attribute__((ext_vector_type(16))) _Float16 v16h;
typedef __attribute__((ext_vector_type(8)))  float    v8f;

#define PHI_F      1.61803398874989484820f
#define INV_2PI_F  0.15915494309189535f
#define TWO_PI_F   6.283185307179586f

// Reduce theta into [0, 2pi) so v_sin_f32 / v_cos_f32 stay in their valid range.
// Mirrors the reference's `theta % TWO_PI`.
__device__ __forceinline__ float wrap2pi(float x) {
    return __builtin_fmaf(-TWO_PI_F, floorf(x * INV_2PI_F), x);
}

// Phase-1 per-element contribution:
//   trig = exp(i*(theta_r)) * exp(i*(theta_i)); return trig_r*xr + trig_i*xi
__device__ __forceinline__ float contrib(float xr, float xi, float wt, float bt, float tphi) {
    const float inv = __builtin_amdgcn_rcpf(1.0f + __builtin_fabsf(wt));
    const float ta  = wrap2pi(__builtin_fmaf(xr, inv, bt + tphi));
    const float tb  = wrap2pi(__builtin_fmaf(xi, inv, bt + tphi));
    const float sa = __sinf(ta), ca = __cosf(ta);
    const float sb = __sinf(tb), cb = __cosf(tb);
    const float tr = ca * cb - sa * sb;
    const float ti = ca * sb + sa * cb;
    return tr * xr + ti * xi;
}

struct TP { float r, i; };

// Phase-2 per-element evolution on alpha-blended state.
__device__ __forceinline__ TP evolve(float xr, float xi, float mr, float mi,
                                     float ws, float bs,
                                     float alpha, float oma, float tphi) {
    const float br  = alpha * xr + oma * mr;
    const float bi  = alpha * xi + oma * mi;
    const float inv = __builtin_amdgcn_rcpf(1.0f + __builtin_fabsf(ws));
    const float ta  = wrap2pi(__builtin_fmaf(br, inv, bs + tphi));
    const float tb  = wrap2pi(__builtin_fmaf(bi, inv, bs + tphi));
    const float sa = __sinf(ta), ca = __cosf(ta);
    const float sb = __sinf(tb), cb = __cosf(tb);
    TP p;
    p.r = ca * cb - sa * sb;
    p.i = ca * sb + sa * cb;
    return p;
}

#define EVOLVE_C(ER, EI, XR, XI, MR, MI, WS, BS, C) do {                      \
    TP p_ = evolve(XR.C, XI.C, MR.C, MI.C, WS.C, BS.C, alpha, oma, tphi);     \
    ER.C = p_.r; EI.C = p_.i; } while (0)

// One workgroup = 16 rows, 512 threads = 16 wave32s.
// Each wave owns 16x32 column tiles (A-fragment layout of v_wmma_f32_16x16x32_f16):
//   lane L: row m = L&15, K-half = L>>4; afrag[i]   = c(kbase+i)      (i=0..7)
//                                        afrag[8+i] = c(16+kbase+i)
// Row-sum via WMMA with all-ones B: D[m,n] = sum_k A[m,k]  (f32 accumulate).
__global__ __launch_bounds__(512)
void liquid_echo_kernel(const float* __restrict__ x_real,
                        const float* __restrict__ x_imag,
                        const float* __restrict__ t_vec,
                        const float* __restrict__ mem_real,
                        const float* __restrict__ mem_imag,
                        const float* __restrict__ w_trigger,
                        const float* __restrict__ b_trigger,
                        const float* __restrict__ w_state,
                        const float* __restrict__ b_state,
                        const float* __restrict__ trigger_bias,
                        float* __restrict__ out_real,
                        float* __restrict__ out_imag,
                        float* __restrict__ out_alpha,
                        int Ddim, float inv_scale)
{
    __shared__ float s_inter[16];
    __shared__ float s_alpha[16];

    const int tid   = threadIdx.x;
    const int lane  = tid & 31;
    const int wave  = tid >> 5;
    const int m     = lane & 15;
    const int khalf = lane >> 4;
    const int kbase = khalf << 3;          // K group base (0 or 8); also M base for C layout
    const int row0  = blockIdx.x << 4;
    const int row   = row0 + m;
    const size_t rowoff = (size_t)row * (size_t)Ddim;
    const int kstep = 16 * 32;             // 16 waves x 32 columns

    if (tid < 16) s_inter[tid] = 0.0f;
    __syncthreads();

    const float tphi = t_vec[row] * PHI_F;

    v16h ones;
    #pragma unroll
    for (int i = 0; i < 16; ++i) ones[i] = (_Float16)1.0f;

    v8f acc = {0.f, 0.f, 0.f, 0.f, 0.f, 0.f, 0.f, 0.f};

    // ---------------- Phase 1: interference row-sums via WMMA ----------------
    for (int k0 = wave * 32; k0 < Ddim; k0 += kstep) {
        v16h afrag;
        #pragma unroll
        for (int g = 0; g < 2; ++g) {
            const int col = k0 + (g << 4) + kbase;
            const float4 xr0 = *(const float4*)(x_real    + rowoff + col);
            const float4 xr1 = *(const float4*)(x_real    + rowoff + col + 4);
            const float4 xi0 = *(const float4*)(x_imag    + rowoff + col);
            const float4 xi1 = *(const float4*)(x_imag    + rowoff + col + 4);
            const float4 wt0 = *(const float4*)(w_trigger + col);
            const float4 wt1 = *(const float4*)(w_trigger + col + 4);
            const float4 bt0 = *(const float4*)(b_trigger + col);
            const float4 bt1 = *(const float4*)(b_trigger + col + 4);
            afrag[(g<<3)+0] = (_Float16)contrib(xr0.x, xi0.x, wt0.x, bt0.x, tphi);
            afrag[(g<<3)+1] = (_Float16)contrib(xr0.y, xi0.y, wt0.y, bt0.y, tphi);
            afrag[(g<<3)+2] = (_Float16)contrib(xr0.z, xi0.z, wt0.z, bt0.z, tphi);
            afrag[(g<<3)+3] = (_Float16)contrib(xr0.w, xi0.w, wt0.w, bt0.w, tphi);
            afrag[(g<<3)+4] = (_Float16)contrib(xr1.x, xi1.x, wt1.x, bt1.x, tphi);
            afrag[(g<<3)+5] = (_Float16)contrib(xr1.y, xi1.y, wt1.y, bt1.y, tphi);
            afrag[(g<<3)+6] = (_Float16)contrib(xr1.z, xi1.z, wt1.z, bt1.z, tphi);
            afrag[(g<<3)+7] = (_Float16)contrib(xr1.w, xi1.w, wt1.w, bt1.w, tphi);
        }
        // D = A * ones + C : every column of D holds the K-tile row sum, f32 accum.
        acc = __builtin_amdgcn_wmma_f32_16x16x32_f16(
                  false, afrag, false, ones, (short)0, acc, false, false);
    }

    // Column N=0 lives in lanes 0 (rows 0..7) and 16 (rows 8..15); combine waves in LDS.
    if ((lane & 15) == 0) {
        #pragma unroll
        for (int j = 0; j < 8; ++j) atomicAdd(&s_inter[kbase + j], acc[j]);
    }
    __syncthreads();

    if (tid < 16) {
        const float z = s_inter[tid] * inv_scale + trigger_bias[0];
        const float a = 1.0f / (1.0f + __expf(-z));
        s_alpha[tid] = a;
        out_alpha[row0 + tid] = a;
    }
    __syncthreads();

    const float alpha = s_alpha[m];
    const float oma   = 1.0f - alpha;

    // ---------------- Phase 2: blend + evolve (x re-reads hit L2) ----------------
    for (int k0 = wave * 32; k0 < Ddim; k0 += kstep) {
        #pragma unroll
        for (int g = 0; g < 2; ++g) {
            const int col = k0 + (g << 4) + kbase;
            const float4 xr0 = *(const float4*)(x_real   + rowoff + col);
            const float4 xr1 = *(const float4*)(x_real   + rowoff + col + 4);
            const float4 xi0 = *(const float4*)(x_imag   + rowoff + col);
            const float4 xi1 = *(const float4*)(x_imag   + rowoff + col + 4);
            const float4 mr0 = *(const float4*)(mem_real + rowoff + col);
            const float4 mr1 = *(const float4*)(mem_real + rowoff + col + 4);
            const float4 mi0 = *(const float4*)(mem_imag + rowoff + col);
            const float4 mi1 = *(const float4*)(mem_imag + rowoff + col + 4);
            const float4 ws0 = *(const float4*)(w_state  + col);
            const float4 ws1 = *(const float4*)(w_state  + col + 4);
            const float4 bs0 = *(const float4*)(b_state  + col);
            const float4 bs1 = *(const float4*)(b_state  + col + 4);

            float4 er0, er1, ei0, ei1;
            EVOLVE_C(er0, ei0, xr0, xi0, mr0, mi0, ws0, bs0, x);
            EVOLVE_C(er0, ei0, xr0, xi0, mr0, mi0, ws0, bs0, y);
            EVOLVE_C(er0, ei0, xr0, xi0, mr0, mi0, ws0, bs0, z);
            EVOLVE_C(er0, ei0, xr0, xi0, mr0, mi0, ws0, bs0, w);
            EVOLVE_C(er1, ei1, xr1, xi1, mr1, mi1, ws1, bs1, x);
            EVOLVE_C(er1, ei1, xr1, xi1, mr1, mi1, ws1, bs1, y);
            EVOLVE_C(er1, ei1, xr1, xi1, mr1, mi1, ws1, bs1, z);
            EVOLVE_C(er1, ei1, xr1, xi1, mr1, mi1, ws1, bs1, w);

            *(float4*)(out_real + rowoff + col)     = er0;
            *(float4*)(out_real + rowoff + col + 4) = er1;
            *(float4*)(out_imag + rowoff + col)     = ei0;
            *(float4*)(out_imag + rowoff + col + 4) = ei1;
        }
    }
}

extern "C" void kernel_launch(void* const* d_in, const int* in_sizes, int n_in,
                              void* d_out, int out_size, void* d_ws, size_t ws_size,
                              hipStream_t stream) {
    const float* x_real = (const float*)d_in[0];
    const float* x_imag = (const float*)d_in[1];
    const float* t_vec  = (const float*)d_in[2];
    const float* mem_r  = (const float*)d_in[3];
    const float* mem_i  = (const float*)d_in[4];
    const float* w_tr   = (const float*)d_in[5];
    const float* b_tr   = (const float*)d_in[6];
    const float* w_st   = (const float*)d_in[7];
    const float* b_st   = (const float*)d_in[8];
    const float* tbias  = (const float*)d_in[9];

    const int B = in_sizes[2];   // length of t
    const int D = in_sizes[5];   // length of w_trigger

    float* out_real  = (float*)d_out;
    float* out_imag  = out_real + (size_t)B * (size_t)D;
    float* out_alpha = out_imag + (size_t)B * (size_t)D;

    const float inv_scale = 1.0f / sqrtf((float)D);

    dim3 grid(B / 16), block(512);
    liquid_echo_kernel<<<grid, block, 0, stream>>>(
        x_real, x_imag, t_vec, mem_r, mem_i,
        w_tr, b_tr, w_st, b_st, tbias,
        out_real, out_imag, out_alpha, D, inv_scale);
}
